// MultiHeadSelfAttention_13219909337792
// MI455X (gfx1250) — matrix operand order repaired and checked by reading, not compiled, not run on hardware
//
#include <hip/hip_runtime.h>

#define DM 768
#define NH 12
#define DK 64
#define BB 4
#define SS 2048
#define MROWS (BB * SS) /* 8192 */

#define AS1 __attribute__((address_space(1)))
#define AS3 __attribute__((address_space(3)))

typedef __attribute__((ext_vector_type(16))) _Float16 v16h;
typedef __attribute__((ext_vector_type(8)))  _Float16 h8;
typedef __attribute__((ext_vector_type(4)))  _Float16 h4;
typedef __attribute__((ext_vector_type(8)))  float    v8f;
typedef __attribute__((ext_vector_type(4)))  int      v4i;

// ---- CDNA5 async global->LDS copy (ASYNCcnt path), 16B per lane ----
static __device__ __forceinline__ void async_cp16(const _Float16* g, _Float16* l) {
#if __has_builtin(__builtin_amdgcn_global_load_async_to_lds_b128)
  __builtin_amdgcn_global_load_async_to_lds_b128(
      (AS1 v4i*)(void*)g, (AS3 v4i*)(void*)l, 0, 0);
#else
  *(h8*)l = *(const h8*)g;  // synchronous fallback
#endif
}

#if __has_builtin(__builtin_amdgcn_s_wait_asynccnt)
#define WAIT_ASYNC(n) __builtin_amdgcn_s_wait_asynccnt(n)
#else
#define WAIT_ASYNC(n) asm volatile("s_wait_asynccnt %0" ::"i"(n) : "memory")
#endif

static __device__ __forceinline__ v16h frag_cat(h8 lo, h8 hi) {
  return __builtin_shufflevector(lo, hi, 0, 1, 2, 3, 4, 5, 6, 7,
                                         8, 9, 10, 11, 12, 13, 14, 15);
}

// A-fragment (16x32 f16). Lanes 0-15: row M=lane, K {0..7,16..23};
// lanes 16-31: row M=lane-16, K {8..15,24..31}.
static __device__ __forceinline__ v16h load_a_frag(const _Float16* base, int stride, int lane) {
  const _Float16* p = base + (lane & 15) * stride + ((lane >> 4) << 3);
  h8 lo = *(const h8*)(p);
  h8 hi = *(const h8*)(p + 16);
  return frag_cat(lo, hi);
}

// B-fragment (32x16 f16). Lane n<16 holds B[K={0..7,16..23}][n], lane n+16 B[K={8..15,24..31}][n].
// Tile stored column-per-row: base + n*stride + k contiguous.
static __device__ __forceinline__ v16h load_b_frag(const _Float16* base, int stride, int lane) {
  const _Float16* p = base + (lane & 15) * stride + ((lane >> 4) << 3);
  h8 lo = *(const h8*)(p);
  h8 hi = *(const h8*)(p + 16);
  return frag_cat(lo, hi);
}

// ---------------------------------------------------------------------------
// Kernel 0a: fp32 -> f16 bulk convert (vectorized x4).
// ---------------------------------------------------------------------------
__global__ __launch_bounds__(256)
void cvt_f16_kernel(const float* __restrict__ src, _Float16* __restrict__ dst, int n4) {
  const int i = blockIdx.x * 256 + threadIdx.x;
  if (i < n4) {
    float4 v = ((const float4*)src)[i];
    h4 o;
    o[0] = (_Float16)v.x; o[1] = (_Float16)v.y;
    o[2] = (_Float16)v.z; o[3] = (_Float16)v.w;
    ((h4*)dst)[i] = o;
  }
}

// ---------------------------------------------------------------------------
// Kernel 0b: W[k][n] fp32 -> Wt[n][k] f16 (transpose-convert, one-time 1.8MB).
// ---------------------------------------------------------------------------
__global__ __launch_bounds__(256)
void wt_f16_kernel(const float* __restrict__ W, _Float16* __restrict__ Wt) {
  const int t = blockIdx.x * 256 + threadIdx.x;  // 0 .. DM*DM-1, coalesced over n
  const int k = t / DM, n = t % DM;
  Wt[(size_t)n * DM + k] = (_Float16)W[t];
}

// ---------------------------------------------------------------------------
// Kernel 1: Y = X @ W + b  (X16 f16 [8192,768], Wt16 f16 [n][k]).
// Block tile 128x128, 8 waves x (64x32). Double-buffered async LDS staging.
// vtrans=0 -> out16 [B,H,S,64]; vtrans=1 -> out16 [B,H,64,S] (for V).
// ---------------------------------------------------------------------------
__global__ __launch_bounds__(256)
void qkv_proj_kernel(const _Float16* __restrict__ X16, const _Float16* __restrict__ Wt16,
                     const float* __restrict__ bias, _Float16* __restrict__ out16,
                     int vtrans) {
  __shared__ _Float16 sA[2][128][40];   // A tile 128x32 (+8 pad)
  __shared__ _Float16 sBt[2][128][40];  // B tile [out-col][k] (+8 pad)

  const int tid  = threadIdx.x;
  const int lane = tid & 31;
  const int wv   = tid >> 5;
  const int wm   = wv >> 2;    // 0..1
  const int wn   = wv & 3;     // 0..3
  const int tn   = blockIdx.x % (DM / 128);
  const int tm   = blockIdx.x / (DM / 128);
  const int m0   = tm * 128;
  const int n0   = tn * 128;

  const v8f vz = {};
  v8f acc[4][2];
#pragma unroll
  for (int i = 0; i < 4; ++i)
#pragma unroll
    for (int j = 0; j < 2; ++j) acc[i][j] = vz;

  // issue both tiles for K-step kk into buffer `buf`: 4 async 16B chunks/thread
  auto issue_ab = [&](int buf, int kk) {
    const int c0 = tid * 2;
#pragma unroll
    for (int u = 0; u < 2; ++u) {
      const int c = c0 + u;
      const int row = c >> 2, seg = (c & 3) << 3;   // 128 rows x 4 chunks
      async_cp16(X16 + (size_t)(m0 + row) * DM + kk + seg, &sA[buf][row][seg]);
      async_cp16(Wt16 + (size_t)(n0 + row) * DM + kk + seg, &sBt[buf][row][seg]);
    }
  };

  issue_ab(0, 0);
  for (int kk = 0; kk < DM; kk += 32) {
    const int cur = (kk >> 5) & 1;
    __syncthreads();  // all reads of buffer cur^1 (prev iter) complete
    if (kk + 32 < DM) {
      issue_ab(cur ^ 1, kk + 32);
      WAIT_ASYNC(4);  // in-order completion: my 4 chunks of tile `cur` landed
    } else {
      WAIT_ASYNC(0);
    }
    __syncthreads();  // every wave passed its wait -> tile `cur` fully in LDS

    v16h afr[4];
#pragma unroll
    for (int im = 0; im < 4; ++im)
      afr[im] = load_a_frag(&sA[cur][wm * 64 + im * 16][0], 40, lane);
#pragma unroll
    for (int in = 0; in < 2; ++in) {
      v16h bfr = load_b_frag(&sBt[cur][wn * 32 + in * 16][0], 40, lane);
#pragma unroll
      for (int im = 0; im < 4; ++im)
        acc[im][in] = __builtin_amdgcn_wmma_f32_16x16x32_f16(
            false, afr[im], false, bfr, (short)0, acc[im][in], false, false);
    }
  }

  // ---- epilogue: + bias, f16 store, head-split (optionally transposed) ----
  const int nlo = lane & 15, rhalf = (lane >> 4) << 3;
#pragma unroll
  for (int im = 0; im < 4; ++im) {
#pragma unroll
    for (int in = 0; in < 2; ++in) {
      const int colg = n0 + wn * 32 + in * 16 + nlo;
      const float bvv = bias[colg];
      const int h = colg >> 6, d = colg & 63;
#pragma unroll
      for (int j = 0; j < 8; ++j) {
        const int row = m0 + wm * 64 + im * 16 + rhalf + j;  // = b*S + s
        const int b = row >> 11, s = row & (SS - 1);
        const _Float16 val = (_Float16)(acc[im][in][j] + bvv);
        if (!vtrans)
          out16[(((size_t)(b * NH + h) * SS) + s) * DK + d] = val;
        else
          out16[(((size_t)(b * NH + h) * DK) + d) * SS + s] = val;
      }
    }
  }
}

// ---------------------------------------------------------------------------
// Kernel 2: flash attention. Block = (b,h, 64-row q tile), 4 waves x 16 rows.
// K tile [key][d] and V tile [d][key] (pre-transposed in global) are staged
// with double-buffered async copies; online softmax in WMMA C-layout.
// ---------------------------------------------------------------------------
__global__ __launch_bounds__(128)
void mha_flash_kernel(const _Float16* __restrict__ q16, const _Float16* __restrict__ k16,
                      const _Float16* __restrict__ v16t, float* __restrict__ out) {
  __shared__ _Float16 sK[2][32][72];    // [key][d] (+pad)
  __shared__ _Float16 sVt[2][64][40];   // [d][key] (+pad)
  __shared__ _Float16 sP[4][16][32];    // per-wave P staging (C->A relayout)

  const int tid = threadIdx.x, lane = tid & 31, wv = tid >> 5;
  const int qt = blockIdx.x & 31;
  const int bh = blockIdx.x >> 5;
  const int b = bh / NH, h = bh % NH;
  const _Float16* qh = q16 + (size_t)bh * SS * DK;
  const _Float16* kh = k16 + (size_t)bh * SS * DK;
  const _Float16* vh = v16t + (size_t)bh * DK * SS;  // [d][s]
  const int qrow0 = qt * 64 + wv * 16;

  // q A-fragments (d 0..31 / 32..63) straight from global
  v16h qf[2];
  {
    const _Float16* p = qh + (size_t)(qrow0 + (lane & 15)) * DK + ((lane >> 4) << 3);
#pragma unroll
    for (int ks = 0; ks < 2; ++ks)
      qf[ks] = frag_cat(*(const h8*)(p + ks * 32), *(const h8*)(p + ks * 32 + 16));
  }

  const v8f vz = {};
  v8f oacc[4];
#pragma unroll
  for (int i = 0; i < 4; ++i) oacc[i] = vz;
  float mrun[8], lrun[8];
#pragma unroll
  for (int j = 0; j < 8; ++j) { mrun[j] = -3.0e38f; lrun[j] = 0.0f; }
  const float scale = 0.125f;  // 1/sqrt(64)

  // stage K (32x64) + V^T (64x32) tiles: 4 async 16B chunks per thread
  auto issue_kv = [&](int buf, int kt) {
    const int c0 = tid * 2;
#pragma unroll
    for (int u = 0; u < 2; ++u) {
      const int c = c0 + u;
      const int kr = c >> 3, kseg = (c & 7) << 3;   // K: 32 rows x 8 chunks
      async_cp16(kh + (size_t)(kt + kr) * DK + kseg, &sK[buf][kr][kseg]);
      const int vr = c >> 2, vseg = (c & 3) << 3;   // V^T: 64 rows x 4 chunks
      async_cp16(vh + (size_t)vr * SS + kt + vseg, &sVt[buf][vr][vseg]);
    }
  };

  issue_kv(0, 0);
  for (int kt = 0; kt < SS; kt += 32) {
    const int cur = (kt >> 5) & 1;
    __syncthreads();
    if (kt + 32 < SS) {
      issue_kv(cur ^ 1, kt + 32);
      WAIT_ASYNC(4);
    } else {
      WAIT_ASYNC(0);
    }
    if (kt + 64 < SS) __builtin_prefetch(kh + (size_t)(kt + 64) * DK, 0, 1);
    __syncthreads();

    // ---- scores: 2 n-tiles (16 keys each) x 2 k-steps over d ----
    v8f sacc[2];
#pragma unroll
    for (int nt = 0; nt < 2; ++nt) {
      v8f z = vz;
#pragma unroll
      for (int ks = 0; ks < 2; ++ks) {
        v16h bf = load_b_frag(&sK[cur][nt * 16][ks * 32], 72, lane);
        z = __builtin_amdgcn_wmma_f32_16x16x32_f16(false, qf[ks], false, bf,
                                                   (short)0, z, false, false);
      }
      sacc[nt] = z;
    }

    // ---- online softmax (element j = row rhalf+j, lanes = keys) ----
    float pv[2][8], alpha[8];
#pragma unroll
    for (int j = 0; j < 8; ++j) {
      float a = sacc[0][j] * scale, c = sacc[1][j] * scale;
      float mx = fmaxf(a, c);
#pragma unroll
      for (int m = 1; m < 16; m <<= 1) mx = fmaxf(mx, __shfl_xor(mx, m, 16));
      const float nm = fmaxf(mrun[j], mx);
      const float p0 = __expf(a - nm);
      const float p1 = __expf(c - nm);
      pv[0][j] = p0; pv[1][j] = p1;
      float s = p0 + p1;
#pragma unroll
      for (int m = 1; m < 16; m <<= 1) s += __shfl_xor(s, m, 16);
      alpha[j] = __expf(mrun[j] - nm);
      lrun[j] = lrun[j] * alpha[j] + s;
      mrun[j] = nm;
    }
#pragma unroll
    for (int dt = 0; dt < 4; ++dt)
#pragma unroll
      for (int j = 0; j < 8; ++j) oacc[dt][j] *= alpha[j];

    // ---- P: C-layout -> LDS -> A-fragment ----
    {
      const int rlo = (lane >> 4) << 3, cl = lane & 15;
#pragma unroll
      for (int nt = 0; nt < 2; ++nt)
#pragma unroll
        for (int j = 0; j < 8; ++j)
          sP[wv][rlo + j][nt * 16 + cl] = (_Float16)pv[nt][j];
    }
    __syncthreads();

    const v16h pf = load_a_frag(&sP[wv][0][0], 32, lane);
#pragma unroll
    for (int dt = 0; dt < 4; ++dt) {
      v16h vf = load_b_frag(&sVt[cur][dt * 16][0], 40, lane);
      oacc[dt] = __builtin_amdgcn_wmma_f32_16x16x32_f16(false, pf, false, vf,
                                                        (short)0, oacc[dt], false, false);
    }
  }

  // ---- epilogue: normalize, merge heads, fp32 [B,S,768] ----
  {
    const int nlo = lane & 15, rhalf = (lane >> 4) << 3;
#pragma unroll
    for (int j = 0; j < 8; ++j) {
      const float inv = 1.0f / lrun[j];
      const int s = qrow0 + rhalf + j;
      float* op = out + ((size_t)b * SS + s) * DM + h * DK;
#pragma unroll
      for (int dt = 0; dt < 4; ++dt) op[dt * 16 + nlo] = oacc[dt][j] * inv;
    }
  }
}

// ---------------------------------------------------------------------------
extern "C" void kernel_launch(void* const* d_in, const int* in_sizes, int n_in,
                              void* d_out, int out_size, void* d_ws, size_t ws_size,
                              hipStream_t stream) {
  (void)in_sizes; (void)n_in; (void)out_size; (void)ws_size;
  const float* X  = (const float*)d_in[0];
  const float* Wq = (const float*)d_in[1];
  const float* bq = (const float*)d_in[2];
  const float* Wk = (const float*)d_in[3];
  const float* bk = (const float*)d_in[4];
  const float* Wv = (const float*)d_in[5];
  const float* bv = (const float*)d_in[6];
  float* out = (float*)d_out;

  const size_t xElems = (size_t)MROWS * DM;   // 6,291,456
  const size_t wElems = (size_t)DM * DM;      //   589,824
  _Float16* X16  = (_Float16*)d_ws;
  _Float16* Wt_q = X16 + xElems;
  _Float16* Wt_k = Wt_q + wElems;
  _Float16* Wt_v = Wt_k + wElems;
  _Float16* q16  = Wt_v + wElems;
  _Float16* k16  = q16 + xElems;
  _Float16* v16t = k16 + xElems;   // total ~54 MB

  cvt_f16_kernel<<<dim3((unsigned)(xElems / 4 / 256)), 256, 0, stream>>>(X, X16, (int)(xElems / 4));
  wt_f16_kernel<<<dim3((unsigned)(wElems / 256)), 256, 0, stream>>>(Wq, Wt_q);
  wt_f16_kernel<<<dim3((unsigned)(wElems / 256)), 256, 0, stream>>>(Wk, Wt_k);
  wt_f16_kernel<<<dim3((unsigned)(wElems / 256)), 256, 0, stream>>>(Wv, Wt_v);

  const dim3 gb((MROWS / 128) * (DM / 128));  // 384 blocks
  qkv_proj_kernel<<<gb, 256, 0, stream>>>(X16, Wt_q, bq, q16, 0);
  qkv_proj_kernel<<<gb, 256, 0, stream>>>(X16, Wt_k, bk, k16, 0);
  qkv_proj_kernel<<<gb, 256, 0, stream>>>(X16, Wt_v, bv, v16t, 1);

  const dim3 ga(BB * NH * (SS / 64));  // 1536 blocks
  mha_flash_kernel<<<ga, 128, 0, stream>>>(q16, k16, v16t, out);
}
